// Linear_48335561949661
// MI455X (gfx1250) — matrix operand order repaired and checked by reading, not compiled, not run on hardware
//
#include <hip/hip_runtime.h>
#include <hip/hip_bf16.h>

typedef __attribute__((ext_vector_type(16))) __bf16 v16bf;
typedef __attribute__((ext_vector_type(8)))  __bf16 v8bf;
typedef __attribute__((ext_vector_type(8)))  float  v8f;
typedef __attribute__((ext_vector_type(4)))  float  v4f;

// Problem dims (fixed by the reference)
#define DIM_M 32      // B*S = 4*8
#define DIM_K 7168    // I
#define DIM_N 18432   // O
#define KBLK  128     // scale block along K
#define NKB   (DIM_K / KBLK)   // 56 K-blocks
#define LDSW  136     // padded LDS row stride in bf16 (272B = 68 banks -> conflict-free)

__device__ __forceinline__ v8bf cvt8(v4f a, v4f b) {
    v8bf r;
    r[0] = (__bf16)a[0];  r[1] = (__bf16)a[1];
    r[2] = (__bf16)a[2];  r[3] = (__bf16)a[3];
    r[4] = (__bf16)b[0];  r[5] = (__bf16)b[1];
    r[6] = (__bf16)b[2];  r[7] = (__bf16)b[3];
    return r;
}

__device__ __forceinline__ v16bf cvt16(v4f a, v4f b, v4f c, v4f d) {
    v16bf r;
    r[0]  = (__bf16)a[0];  r[1]  = (__bf16)a[1];
    r[2]  = (__bf16)a[2];  r[3]  = (__bf16)a[3];
    r[4]  = (__bf16)b[0];  r[5]  = (__bf16)b[1];
    r[6]  = (__bf16)b[2];  r[7]  = (__bf16)b[3];
    r[8]  = (__bf16)c[0];  r[9]  = (__bf16)c[1];
    r[10] = (__bf16)c[2];  r[11] = (__bf16)c[3];
    r[12] = (__bf16)d[0];  r[13] = (__bf16)d[1];
    r[14] = (__bf16)d[2];  r[15] = (__bf16)d[3];
    return r;
}

__device__ __forceinline__ v16bf comb(v8bf a, v8bf b) {
    v16bf r;
#pragma unroll
    for (int i = 0; i < 8; ++i) { r[i] = a[i]; r[i + 8] = b[i]; }
    return r;
}

// One A fragment (16 bf16 / lane) = two ds_load_b128 from the staged block.
__device__ __forceinline__ v16bf loadA(const __bf16* ab, int row, int koff) {
    const __bf16* p = ab + row * LDSW + koff;
    return comb(*(const v8bf*)(p), *(const v8bf*)(p + 8));
}

// Cooperatively stage one 32x128 f32 block of x into LDS as bf16.
// 256 threads: thread t handles row t/8, 16-float segment t%8.
__device__ __forceinline__ void stage_x(const float* __restrict__ x,
                                        __bf16* lds, int k0, int tid) {
    const int row = tid >> 3;
    const int seg = tid & 7;
    const v4f* p = (const v4f*)(x + (size_t)row * DIM_K + k0 + seg * 16);
    v4f f0 = p[0], f1 = p[1], f2 = p[2], f3 = p[3];
    __bf16* dst = lds + row * LDSW + seg * 16;
    *(v8bf*)(dst)     = cvt8(f0, f1);
    *(v8bf*)(dst + 8) = cvt8(f2, f3);
}

__global__ __launch_bounds__(256) void
linear_bf16_wmma_kernel(const float* __restrict__ x,      // [32, 7168]
                        const float* __restrict__ w,      // [18432, 7168]
                        const float* __restrict__ scale,  // [144, 56]
                        float* __restrict__ y)            // [32, 18432]
{
    __shared__ __attribute__((aligned(16))) __bf16 abuf[2][DIM_M * LDSW]; // 2 x 8704 B

    const int tid  = threadIdx.x;
    const int lane = tid & 31;
    const int wave = tid >> 5;
    const int lo   = lane & 15;     // N-column within tile / M-row selector
    const int half = lane >> 4;     // K-half selector (contiguous k = 16*half + s)

    const int tileN = (blockIdx.x * 8 + wave) * 16;   // 16 output columns per wave

    // Weight stream base (B fragment: lanes 0-15 K={k..k+15},
    // lanes 16-31 K={k+16..k+31}, column = lane%16).
    const float* wr = w + (size_t)(tileN + lo) * DIM_K + half * 16;

    // A fragment K sub-offset within a 32-K chunk (contiguous k order, same as B):
    //   lanes 0-15: {k+0..15}; lanes 16-31: {k+16..31}
    const int aoff = half * 16;

    const int srow = (tileN >> 7) * NKB;  // scale row for this column block

    v8f c0 = {};   // rows 0-15
    v8f c1 = {};   // rows 16-31

    // Preload first x block into buffer 0.
    stage_x(x, abuf[0], 0, tid);
    __syncthreads();

    for (int kb = 0; kb < NKB; ++kb) {
        const __bf16* ab = abuf[kb & 1];
        const int k0 = kb * KBLK;
        const float s = scale[srow + kb];        // wave-uniform -> s_load

        if (kb + 1 < NKB)
            __builtin_prefetch(wr + k0 + KBLK, 0, 1);

        v8f b0 = {};
        v8f b1 = {};

        // Software pipeline: A fragments for chunk kc+1 are issued before
        // the WMMAs of chunk kc, giving the ds_loads latency cover.
        v16bf A0 = loadA(ab, lo,      aoff);
        v16bf A1 = loadA(ab, 16 + lo, aoff);

#pragma unroll
        for (int kc = 0; kc < 4; ++kc) {
            const int k = kc * 32;

            v16bf A0n, A1n;
            if (kc < 3) {
                A0n = loadA(ab, lo,      k + 32 + aoff);
                A1n = loadA(ab, 16 + lo, k + 32 + aoff);
            }

            // B fragment: 16 contiguous f32 from the weight stream.
            const v4f* pb = (const v4f*)(wr + k0 + k);
            v16bf B = cvt16(pb[0], pb[1], pb[2], pb[3]);

            // D = A x B + C (f32 accumulate)
            b0 = __builtin_amdgcn_wmma_f32_16x16x32_bf16(
                     false, A0, false, B, (short)0, b0, false, false);
            b1 = __builtin_amdgcn_wmma_f32_16x16x32_bf16(
                     false, A1, false, B, (short)0, b1, false, false);

            if (kc < 3) { A0 = A0n; A1 = A1n; }
        }

        // Fold the block scale once per 128-K block.
#pragma unroll
        for (int i = 0; i < 8; ++i) {
            c0[i] += s * b0[i];
            c1[i] += s * b1[i];
        }

        // Stage next x block into the other buffer (overlaps next compute).
        if (kb + 1 < NKB)
            stage_x(x, abuf[(kb + 1) & 1], (kb + 1) * KBLK, tid);
        __syncthreads();
    }

    // D layout (32-bit C/D 16x16): lane%16 = column, VGPR i = row i + 8*(lane/16)
    {
        float* yp = y + (size_t)(half * 8) * DIM_N + (tileN + lo);
#pragma unroll
        for (int i = 0; i < 8; ++i)
            yp[(size_t)i * DIM_N] = c0[i];
    }
    {
        float* yp = y + (size_t)(16 + half * 8) * DIM_N + (tileN + lo);
#pragma unroll
        for (int i = 0; i < 8; ++i)
            yp[(size_t)i * DIM_N] = c1[i];
    }
}

extern "C" void kernel_launch(void* const* d_in, const int* in_sizes, int n_in,
                              void* d_out, int out_size, void* d_ws, size_t ws_size,
                              hipStream_t stream) {
    const float* x     = (const float*)d_in[0];   // [4,8,7168] f32
    const float* wgt   = (const float*)d_in[1];   // [18432,7168] f32
    const float* scale = (const float*)d_in[2];   // [144,56] f32
    float* y           = (float*)d_out;           // [4,8,18432] f32

    (void)in_sizes; (void)n_in; (void)out_size; (void)d_ws; (void)ws_size;

    dim3 grid(DIM_N / (8 * 16));   // 144 blocks, 8 waves/block, 16 cols/wave
    dim3 block(256);
    linear_bf16_wmma_kernel<<<grid, block, 0, stream>>>(x, wgt, scale, y);
}
